// GraphConvolutionalNetwork_37271726195316
// MI455X (gfx1250) — matrix operand order repaired and checked by reading, not compiled, not run on hardware
//
#include <hip/hip_runtime.h>

// ---------------------------------------------------------------------------
// GCN (2-layer, PyG GCNConv semantics) for MI455X / gfx1250.
// GEMMs via v_wmma_f32_16x16x32_bf16 (wave32, one wave per 16x16 C tile).
// Edge aggregation via f32 L2 atomics; x_lin (51.2MB) is L2-resident (192MB).
// ---------------------------------------------------------------------------

typedef __attribute__((ext_vector_type(16))) __bf16 v16bf;
typedef __attribute__((ext_vector_type(8)))  __bf16 v8bf;
typedef __attribute__((ext_vector_type(8)))  float  v8f;

__device__ __forceinline__ unsigned short f32_to_bf16_rne(float f) {
  unsigned int u = __float_as_uint(f);
  u += 0x7FFFu + ((u >> 16) & 1u);          // round-to-nearest-even
  return (unsigned short)(u >> 16);
}

// ------------------------------- utility -----------------------------------

__global__ void zero_f32_kernel(float* __restrict__ p, int n) {
  int i = blockIdx.x * blockDim.x + threadIdx.x;
  if (i < n) p[i] = 0.0f;
}

__global__ void deg_count_kernel(const int* __restrict__ dst,
                                 float* __restrict__ deg, int E) {
  int e = blockIdx.x * blockDim.x + threadIdx.x;
  if (e < E) atomicAdd(&deg[dst[e]], 1.0f);
}

__global__ void dinv_sqrt_kernel(float* __restrict__ deg, int n) {
  int i = blockIdx.x * blockDim.x + threadIdx.x;
  if (i < n) deg[i] = rsqrtf(deg[i] + 1.0f);   // includes self-loop
}

__global__ void f32_to_bf16_kernel(const float* __restrict__ x,
                                   unsigned short* __restrict__ y, int n) {
  int i = blockIdx.x * blockDim.x + threadIdx.x;
  if (i < n) y[i] = f32_to_bf16_rne(x[i]);
}

// W is K x N (row-major, in x out). Emit WT = N x K bf16 so a B-matrix column
// is a contiguous run of K (matches per-lane contiguous B fragment loads).
__global__ void wT_bf16_kernel(const float* __restrict__ W,
                               unsigned short* __restrict__ WT, int K, int N) {
  int i = blockIdx.x * blockDim.x + threadIdx.x;
  if (i < K * N) {
    int k = i / N, n = i - k * N;
    WT[n * K + k] = f32_to_bf16_rne(W[i]);
  }
}

// ------------------------------- WMMA GEMM ---------------------------------
// C[M,N] = A[M,K] * B[K,N], A row-major bf16, B given as BT[N,K] bf16.
// One wave per 16x16 C tile; K-loop in steps of 32 (8 WMMAs for K=256).
// A-fragment layout: lane<16 -> row=lane, K=k..k+15;
// lane>=16 -> row=lane-16, K=k+16..k+31 (same k order as B).
// B-fragment layout (contiguous k order, same as A): lane<16 -> col=lane, K=k..k+15;
// lane>=16 -> col=lane-16, K=k+16..k+31 (contiguous per lane thanks to BT).
__global__ void gemm_bf16_wmma_kernel(const unsigned short* __restrict__ A,
                                      const unsigned short* __restrict__ BT,
                                      float* __restrict__ C,
                                      int M, int N, int K) {
  const int lane   = threadIdx.x & 31;
  const int wave   = threadIdx.x >> 5;
  const int tilesN = N >> 4;
  const int tile   = blockIdx.x * (blockDim.x >> 5) + wave;
  if (tile >= (M >> 4) * tilesN) return;     // wave-uniform exit

  const int tM  = tile / tilesN;
  const int tN  = tile - tM * tilesN;
  const int r16 = lane & 15;
  const int sel = lane >> 4;                 // 0: low half, 1: high half

  const unsigned short* __restrict__ arow = A  + (long long)(tM * 16 + r16) * K;
  const unsigned short* __restrict__ brow = BT + (long long)(tN * 16 + r16) * K;

  v8f acc = {0.f, 0.f, 0.f, 0.f, 0.f, 0.f, 0.f, 0.f};

  for (int k = 0; k < K; k += 32) {
    v16bf a = *(const v16bf*)(arow + k + sel * 16);       // 32B load
    v16bf b = *(const v16bf*)(brow + k + sel * 16);       // 32B load
    acc = __builtin_amdgcn_wmma_f32_16x16x32_bf16(
        /*neg_a=*/false, a, /*neg_b=*/false, b,
        /*c_mod=*/(short)0, acc, /*reuse_a=*/false, /*reuse_b=*/false);
  }

  // C/D layout: VGPR r -> row r (lanes 0-15) / row r+8 (lanes 16-31), N=lane%16
  float* __restrict__ crow = C + (long long)(tM * 16 + sel * 8) * N + tN * 16 + r16;
  #pragma unroll
  for (int r = 0; r < 8; ++r) crow[(long long)r * N] = acc[r];
}

// ----------------------------- edge scatter --------------------------------
// One thread per (edge, 4 contiguous cols): float4 gather + 4 f32 atomics.
__global__ void scatter_kernel(const float* __restrict__ xlin,
                               const int* __restrict__ src,
                               const int* __restrict__ dst,
                               const float* __restrict__ dinv,
                               float* __restrict__ agg, int E, int C) {
  long long tid = (long long)blockIdx.x * blockDim.x + threadIdx.x;
  const int chunks = C >> 2;
  if (tid >= (long long)E * chunks) return;
  int e  = (int)(tid / chunks);
  int c4 = ((int)(tid - (long long)e * chunks)) << 2;
  int s = src[e], d = dst[e];
  float coef = dinv[s] * dinv[d];
  const float4 v = *(const float4*)(xlin + (long long)s * C + c4);
  float* o = agg + (long long)d * C + c4;
  atomicAdd(o + 0, v.x * coef);
  atomicAdd(o + 1, v.y * coef);
  atomicAdd(o + 2, v.z * coef);
  atomicAdd(o + 3, v.w * coef);
}

// h = relu(agg + xlin/deg + b); emit bf16 for next layer's WMMA A matrix.
__global__ void selfloop_relu_bf16_kernel(const float* __restrict__ agg,
                                          const float* __restrict__ xlin,
                                          const float* __restrict__ dinv,
                                          const float* __restrict__ b,
                                          unsigned short* __restrict__ h_bf16,
                                          int M, int C) {
  int i = blockIdx.x * blockDim.x + threadIdx.x;
  if (i >= M * C) return;
  int row = i / C, c = i - row * C;
  float di = dinv[row];
  float v = agg[i] + xlin[i] * di * di + b[c];
  h_bf16[i] = f32_to_bf16_rne(fmaxf(v, 0.0f));
}

// out += xlin/deg + b  (out already holds layer-2 aggregation)
__global__ void selfloop_out_kernel(const float* __restrict__ xlin,
                                    const float* __restrict__ dinv,
                                    const float* __restrict__ b,
                                    float* __restrict__ out, int M, int C) {
  int i = blockIdx.x * blockDim.x + threadIdx.x;
  if (i >= M * C) return;
  int row = i / C, c = i - row * C;
  float di = dinv[row];
  out[i] += xlin[i] * di * di + b[c];
}

// ------------------------------- launcher ----------------------------------

static inline int cdiv(long long a, long long b) { return (int)((a + b - 1) / b); }

extern "C" void kernel_launch(void* const* d_in, const int* in_sizes, int n_in,
                              void* d_out, int out_size, void* d_ws, size_t ws_size,
                              hipStream_t stream) {
  const float* x  = (const float*)d_in[0];
  const int*   ei = (const int*)d_in[1];
  const float* W1 = (const float*)d_in[2];
  const float* b1 = (const float*)d_in[3];
  const float* W2 = (const float*)d_in[4];
  const float* b2 = (const float*)d_in[5];
  float* out = (float*)d_out;

  const int IN_C = 256, HID_C = 256, OUT_C = 128;
  const int M = in_sizes[0] / IN_C;     // 50000 (multiple of 16)
  const int E = in_sizes[1] / 2;        // 800000
  const int* src = ei;
  const int* dst = ei + E;

  // workspace carve-up (256B aligned)
  char* ws = (char*)d_ws;
  size_t off = 0;
  auto take = [&](size_t bytes) -> char* {
    char* p = ws + off;
    off = (off + bytes + 255) & ~(size_t)255;
    return p;
  };
  float*          dinv = (float*)take((size_t)M * 4);
  unsigned short* W1T  = (unsigned short*)take((size_t)IN_C * HID_C * 2);
  unsigned short* W2T  = (unsigned short*)take((size_t)HID_C * OUT_C * 2);
  unsigned short* xbf  = (unsigned short*)take((size_t)M * IN_C * 2);  // reused as h_bf16
  float*          xlin = (float*)take((size_t)M * HID_C * 4);          // reused layer 2
  float*          agg  = (float*)take((size_t)M * HID_C * 4);
  (void)ws_size; (void)n_in; (void)out_size;

  const int T = 256;

  // degrees -> dinv
  zero_f32_kernel<<<cdiv(M, T), T, 0, stream>>>(dinv, M);
  deg_count_kernel<<<cdiv(E, T), T, 0, stream>>>(dst, dinv, E);
  dinv_sqrt_kernel<<<cdiv(M, T), T, 0, stream>>>(dinv, M);

  // bf16 conversions
  f32_to_bf16_kernel<<<cdiv((long long)M * IN_C, T), T, 0, stream>>>(x, xbf, M * IN_C);
  wT_bf16_kernel<<<cdiv(IN_C * HID_C, T), T, 0, stream>>>(W1, W1T, IN_C, HID_C);
  wT_bf16_kernel<<<cdiv(HID_C * OUT_C, T), T, 0, stream>>>(W2, W2T, HID_C, OUT_C);

  // layer 1: x_lin = x @ W1 (WMMA), aggregate, self-loop + bias + relu
  {
    int tiles = (M >> 4) * (HID_C >> 4);
    gemm_bf16_wmma_kernel<<<cdiv(tiles, T / 32), T, 0, stream>>>(xbf, W1T, xlin, M, HID_C, IN_C);
    zero_f32_kernel<<<cdiv((long long)M * HID_C, T), T, 0, stream>>>(agg, M * HID_C);
    scatter_kernel<<<cdiv((long long)E * (HID_C / 4), T), T, 0, stream>>>(xlin, src, dst, dinv, agg, E, HID_C);
    selfloop_relu_bf16_kernel<<<cdiv((long long)M * HID_C, T), T, 0, stream>>>(agg, xlin, dinv, b1, xbf, M, HID_C);
  }

  // layer 2: x_lin2 = h @ W2 (WMMA), aggregate into d_out, self-loop + bias
  {
    int tiles = (M >> 4) * (OUT_C >> 4);
    gemm_bf16_wmma_kernel<<<cdiv(tiles, T / 32), T, 0, stream>>>(xbf, W2T, xlin, M, OUT_C, HID_C);
    zero_f32_kernel<<<cdiv((long long)M * OUT_C, T), T, 0, stream>>>(out, M * OUT_C);
    scatter_kernel<<<cdiv((long long)E * (OUT_C / 4), T), T, 0, stream>>>(xlin, src, dst, dinv, out, E, OUT_C);
    selfloop_out_kernel<<<cdiv((long long)M * OUT_C, T), T, 0, stream>>>(xlin, dinv, b2, out, M, OUT_C);
  }
}
